// ChannelAttention_7103875907887
// MI455X (gfx1250) — matrix operand order repaired and checked by reading, not compiled, not run on hardware
//
#include <hip/hip_runtime.h>
#include <hip/hip_bf16.h>
#include <math.h>

typedef __attribute__((ext_vector_type(16))) __bf16        v16bf;
typedef __attribute__((ext_vector_type(8)))  float         v8f;
typedef __attribute__((ext_vector_type(4)))  unsigned int  v4u;

#define B_N    8
#define C_N    192
#define C3_N   576
#define HGT    128
#define WID    128
#define NPIX   16384
#define NHEADS 8
#define CHEAD  24
#define GSEG   8          // K-split segments for the Gram reduction

__device__ __forceinline__ unsigned short f2bf(float f) {
  unsigned int u = __builtin_bit_cast(unsigned int, f);
  u += 0x7FFFu + ((u >> 16) & 1u);            // round-to-nearest-even
  return (unsigned short)(u >> 16);
}
__device__ __forceinline__ float bf2f(unsigned short h) {
  unsigned int u = ((unsigned int)h) << 16;
  return __builtin_bit_cast(float, u);
}

union BF16Frag { v16bf v; unsigned short u[16]; v4u q[2]; };

__device__ __forceinline__ v4u pack8bf(const v8f& a) {
  v4u r;
  r.x = (unsigned)f2bf(a[0]) | ((unsigned)f2bf(a[1]) << 16);
  r.y = (unsigned)f2bf(a[2]) | ((unsigned)f2bf(a[3]) << 16);
  r.z = (unsigned)f2bf(a[4]) | ((unsigned)f2bf(a[5]) << 16);
  r.w = (unsigned)f2bf(a[6]) | ((unsigned)f2bf(a[7]) << 16);
  return r;
}

// ---------------------------------------------------------------------------
// Kernel 1: depthwise 3x3 on x, unbiased variance over H,W -> var[b*C + c]
// ---------------------------------------------------------------------------
__global__ void k_dwvar(const float* __restrict__ x, const float* __restrict__ wdw,
                        float* __restrict__ var) {
  __shared__ float red[256];
  const int bc = blockIdx.x;             // b*C + c
  const int c = bc % C_N;
  const float* xp = x + (size_t)bc * NPIX;
  float w[9];
#pragma unroll
  for (int i = 0; i < 9; ++i) w[i] = wdw[c * 9 + i];
  float s1 = 0.f, s2 = 0.f;
  for (int i = threadIdx.x; i < NPIX; i += 256) {
    int y = i >> 7, xx = i & 127;
    float a = 0.f;
#pragma unroll
    for (int dy = -1; dy <= 1; ++dy) {
      int yy = y + dy; if ((unsigned)yy >= (unsigned)HGT) continue;
#pragma unroll
      for (int dx = -1; dx <= 1; ++dx) {
        int xc = xx + dx; if ((unsigned)xc >= (unsigned)WID) continue;
        a += xp[yy * WID + xc] * w[(dy + 1) * 3 + (dx + 1)];
      }
    }
    s1 += a; s2 += a * a;
  }
  red[threadIdx.x] = s1; __syncthreads();
  for (int s = 128; s > 0; s >>= 1) { if (threadIdx.x < s) red[threadIdx.x] += red[threadIdx.x + s]; __syncthreads(); }
  float t1 = red[0]; __syncthreads();
  red[threadIdx.x] = s2; __syncthreads();
  for (int s = 128; s > 0; s >>= 1) { if (threadIdx.x < s) red[threadIdx.x] += red[threadIdx.x + s]; __syncthreads(); }
  if (threadIdx.x == 0) {
    float t2 = red[0];
    var[bc] = (t2 - t1 * t1 / (float)NPIX) / (float)(NPIX - 1);
  }
}

// ---------------------------------------------------------------------------
// Kernel 2: qkv 1x1-conv GEMM  Y[b,m,n] = sum_k W[m,k] * X[b,k,n], f32 -> bf16
// BM=64, BN=128, BK=32, 256 threads (8 waves), 2x2 WMMA tiles per wave.
// B tile stored transposed in LDS so fragments are contiguous b128 loads.
// ---------------------------------------------------------------------------
template <typename TIN, typename TOUT>
__global__ void k_gemm_wmma(const float* __restrict__ Wt, const TIN* __restrict__ Xin,
                            TOUT* __restrict__ Yout, int M, int K) {
  constexpr int BM = 64, BN = 128, BK = 32;
  __shared__ __align__(16) unsigned short As[BM][BK + 8];   // [m][k], stride 80B
  __shared__ __align__(16) unsigned short Bs[BN][BK + 8];   // [n][k], stride 80B
  const int b = blockIdx.z;
  const TIN* X = Xin + (size_t)b * K * NPIX;
  TOUT* Y = Yout + (size_t)b * M * NPIX;
  const int m_base = blockIdx.y * BM;
  const int n_base = blockIdx.x * BN;
  const int tid = threadIdx.x;
  const int wave = tid >> 5, lane = tid & 31;
  const int wm = (wave >> 2) * 32;       // 0 or 32
  const int wn = (wave & 3) * 32;        // 0,32,64,96
  const int lhi = lane >> 4, llo = lane & 15;

  const v8f vz = {0.f,0.f,0.f,0.f,0.f,0.f,0.f,0.f};
  v8f acc[2][2];
  acc[0][0] = vz; acc[0][1] = vz; acc[1][0] = vz; acc[1][1] = vz;

  for (int kb = 0; kb < K; kb += BK) {
    for (int j = tid; j < BM * BK; j += 256) {
      int r = j >> 5, cc = j & 31;
      As[r][cc] = f2bf(Wt[(size_t)(m_base + r) * K + kb + cc]);
    }
    for (int j = tid; j < BK * BN; j += 256) {
      int r = j >> 7, cc = j & 127;          // r = k, cc = n (coalesced on n)
      TIN vsrc = X[(size_t)(kb + r) * NPIX + n_base + cc];
      if constexpr (sizeof(TIN) == 4) Bs[cc][r] = f2bf((float)vsrc);
      else                            Bs[cc][r] = (unsigned short)vsrc;
    }
    if (kb + BK < K)  // warm caches for next K panel (global_prefetch_b8)
      __builtin_prefetch((const void*)&X[(size_t)(kb + BK + (tid >> 7)) * NPIX + n_base + (tid & 127)], 0, 1);
    __syncthreads();

    BF16Frag a[2], bb[2];
#pragma unroll
    for (int t = 0; t < 2; ++t) {
      const int mrow = wm + t * 16 + llo;
      a[t].q[0] = *reinterpret_cast<const v4u*>(&As[mrow][lhi * 16]);
      a[t].q[1] = *reinterpret_cast<const v4u*>(&As[mrow][lhi * 16 + 8]);
      const int ncol = wn + t * 16 + llo;
      const v4u* bp = reinterpret_cast<const v4u*>(&Bs[ncol][lhi * 16]);
      bb[t].q[0] = bp[0];
      bb[t].q[1] = bp[1];
    }
#pragma unroll
    for (int i = 0; i < 2; ++i)
#pragma unroll
      for (int j = 0; j < 2; ++j)
        acc[i][j] = __builtin_amdgcn_wmma_f32_16x16x32_bf16(
            false, a[i].v, false, bb[j].v, (short)0, acc[i][j], false, false);
    __syncthreads();
  }
#pragma unroll
  for (int i = 0; i < 2; ++i)
#pragma unroll
    for (int j = 0; j < 2; ++j)
#pragma unroll
      for (int r = 0; r < 8; ++r) {
        int m = m_base + wm + i * 16 + r + 8 * lhi;
        int n = n_base + wn + j * 16 + llo;
        float v = acc[i][j][r];
        if constexpr (sizeof(TOUT) == 4) Y[(size_t)m * NPIX + n] = v;
        else                             Y[(size_t)m * NPIX + n] = f2bf(v);
      }
}

// ---------------------------------------------------------------------------
// Kernel 3: on-the-fly dw-conv of q/k channels, L2-norm inverses
// ---------------------------------------------------------------------------
__global__ void k_norm(const unsigned short* __restrict__ qkv0,
                       const float* __restrict__ wqd,
                       float* __restrict__ invq, float* __restrict__ invk) {
  __shared__ float red[256];
  const int blk = blockIdx.x;                 // b*(2C) + j
  const int b = blk / (2 * C_N);
  const int j = blk % (2 * C_N);              // q: 0..191, k: 192..383
  const unsigned short* plane = qkv0 + ((size_t)b * C3_N + j) * NPIX;
  float w[9];
#pragma unroll
  for (int i = 0; i < 9; ++i) w[i] = wqd[j * 9 + i];
  float s2 = 0.f;
  for (int i = threadIdx.x; i < NPIX; i += 256) {
    int y = i >> 7, xx = i & 127;
    float a = 0.f;
#pragma unroll
    for (int dy = -1; dy <= 1; ++dy) {
      int yy = y + dy; if ((unsigned)yy >= (unsigned)HGT) continue;
#pragma unroll
      for (int dx = -1; dx <= 1; ++dx) {
        int xc = xx + dx; if ((unsigned)xc >= (unsigned)WID) continue;
        a += bf2f(plane[yy * WID + xc]) * w[(dy + 1) * 3 + (dx + 1)];
      }
    }
    s2 += a * a;
  }
  red[threadIdx.x] = s2; __syncthreads();
  for (int s = 128; s > 0; s >>= 1) { if (threadIdx.x < s) red[threadIdx.x] += red[threadIdx.x + s]; __syncthreads(); }
  if (threadIdx.x == 0) {
    float inv = 1.0f / fmaxf(sqrtf(red[0]), 1e-12f);
    if (j < C_N) invq[b * C_N + j] = inv;
    else         invk[b * C_N + (j - C_N)] = inv;
  }
}

// ---------------------------------------------------------------------------
// Kernel 4: partial Gram q·k^T via WMMA. 512 blocks: (b,head) x 8 K-segments
// of 2048 pixels. Each block writes a deterministic partial 32x32 sum.
// ---------------------------------------------------------------------------
__global__ void k_gram_part(const unsigned short* __restrict__ qkv0,
                            const float* __restrict__ wqd,
                            float* __restrict__ gram_part) {
  constexpr int NC = 256, LSTR = NC + 8;      // 528 B row stride (16B aligned)
  __shared__ __align__(16) unsigned short qs[CHEAD][LSTR];
  __shared__ __align__(16) unsigned short ks[CHEAD][LSTR];
  __shared__ float red[8 * 32 * 32];
  const int blk = blockIdx.x;
  const int seg = blk % GSEG;
  const int bh = blk / GSEG;
  const int b = bh / NHEADS, h = bh % NHEADS;
  const int tid = threadIdx.x;
  const int wave = tid >> 5, lane = tid & 31;
  const int lhi = lane >> 4, llo = lane & 15;

  const v8f vz = {0.f,0.f,0.f,0.f,0.f,0.f,0.f,0.f};
  const v4u zz = {0u,0u,0u,0u};
  v8f acc[2][2];
  acc[0][0] = vz; acc[0][1] = vz; acc[1][0] = vz; acc[1][1] = vz;

  for (int chunk = seg * (NPIX / NC / GSEG); chunk < (seg + 1) * (NPIX / NC / GSEG); ++chunk) {
    const int n = chunk * NC + tid;
    const int y = n >> 7, xx = n & 127;
    for (int c = 0; c < CHEAD; ++c) {
      const int qc = h * CHEAD + c;
      const unsigned short* qp = qkv0 + ((size_t)b * C3_N + qc) * NPIX;
      const unsigned short* kp = qkv0 + ((size_t)b * C3_N + C_N + qc) * NPIX;
      const float* wq = wqd + qc * 9;
      const float* wk = wqd + (C_N + qc) * 9;
      float aq = 0.f, ak = 0.f;
#pragma unroll
      for (int dy = -1; dy <= 1; ++dy) {
        int yy = y + dy; if ((unsigned)yy >= (unsigned)HGT) continue;
#pragma unroll
        for (int dx = -1; dx <= 1; ++dx) {
          int xc = xx + dx; if ((unsigned)xc >= (unsigned)WID) continue;
          int o = yy * WID + xc, wi = (dy + 1) * 3 + (dx + 1);
          aq += bf2f(qp[o]) * wq[wi];
          ak += bf2f(kp[o]) * wk[wi];
        }
      }
      qs[c][tid] = f2bf(aq);
      ks[c][tid] = f2bf(ak);
    }
    __syncthreads();

    BF16Frag a[2], bb[2];
#pragma unroll
    for (int t = 0; t < 2; ++t) {
      const int m = t * 16 + llo;               // Gram row (q channel)
      if (m < CHEAD) {
        a[t].q[0] = *reinterpret_cast<const v4u*>(&qs[m][wave * 32 + lhi * 16]);
        a[t].q[1] = *reinterpret_cast<const v4u*>(&qs[m][wave * 32 + lhi * 16 + 8]);
      } else { a[t].q[0] = zz; a[t].q[1] = zz; }
      const int d = t * 16 + llo;               // Gram col (k channel)
      if (d < CHEAD) {
        const v4u* bp = reinterpret_cast<const v4u*>(&ks[d][wave * 32 + lhi * 16]);
        bb[t].q[0] = bp[0];
        bb[t].q[1] = bp[1];
      } else { bb[t].q[0] = zz; bb[t].q[1] = zz; }
    }
#pragma unroll
    for (int i = 0; i < 2; ++i)
#pragma unroll
      for (int j = 0; j < 2; ++j)
        acc[i][j] = __builtin_amdgcn_wmma_f32_16x16x32_bf16(
            false, a[i].v, false, bb[j].v, (short)0, acc[i][j], false, false);
    __syncthreads();
  }

  // cross-wave reduce of the 32x32 partial Gram, then one global write
#pragma unroll
  for (int i = 0; i < 2; ++i)
#pragma unroll
    for (int j = 0; j < 2; ++j)
#pragma unroll
      for (int r = 0; r < 8; ++r) {
        int m = i * 16 + r + 8 * lhi;
        int d = j * 16 + llo;
        red[wave * 1024 + m * 32 + d] = acc[i][j][r];
      }
  __syncthreads();
  float* out = gram_part + (size_t)(bh * GSEG + seg) * 1024;
  for (int idx = tid; idx < 1024; idx += 256) {
    float s = 0.f;
#pragma unroll
    for (int w = 0; w < 8; ++w) s += red[w * 1024 + idx];
    out[idx] = s;
  }
}

// ---------------------------------------------------------------------------
// Kernel 5: reduce Gram partials, scale by inv-norms*temperature, add
// diagonal rescale*var, row-softmax -> zero-padded 32x32 bf16 attn.
// ---------------------------------------------------------------------------
__global__ void k_softmax(const float* __restrict__ gram_part,
                          const float* __restrict__ invq, const float* __restrict__ invk,
                          const float* __restrict__ temperature,
                          const float* __restrict__ rescale,
                          const float* __restrict__ var,
                          unsigned short* __restrict__ attn) {
  __shared__ float Gm[1024];
  const int bh = blockIdx.x;
  const int b = bh / NHEADS, h = bh % NHEADS;
  const int tid = threadIdx.x;
  for (int idx = tid; idx < 1024; idx += 256) {
    float s = 0.f;
#pragma unroll
    for (int g = 0; g < GSEG; ++g) s += gram_part[(size_t)(bh * GSEG + g) * 1024 + idx];
    int m = idx >> 5, d = idx & 31;
    float val = 0.f;
    if (m < CHEAD && d < CHEAD) {
      val = s * invq[b * C_N + h * CHEAD + m] * invk[b * C_N + h * CHEAD + d] * temperature[h];
      if (m == d) val += rescale[h] * var[b * C_N + h * CHEAD + m];
    }
    Gm[idx] = val;
  }
  __syncthreads();
  if (tid < 32) {
    size_t base = (size_t)bh * 1024 + (size_t)tid * 32;
    if (tid < CHEAD) {
      float mx = -1e30f;
#pragma unroll
      for (int d = 0; d < CHEAD; ++d) mx = fmaxf(mx, Gm[tid * 32 + d]);
      float ex[CHEAD]; float s = 0.f;
#pragma unroll
      for (int d = 0; d < CHEAD; ++d) { ex[d] = __expf(Gm[tid * 32 + d] - mx); s += ex[d]; }
      float inv = 1.0f / s;
#pragma unroll
      for (int d = 0; d < CHEAD; ++d) attn[base + d] = f2bf(ex[d] * inv);
#pragma unroll
      for (int d = CHEAD; d < 32; ++d) attn[base + d] = 0;
    } else {
      for (int d = 0; d < 32; ++d) attn[base + d] = 0;
    }
  }
}

// ---------------------------------------------------------------------------
// Kernel 6: out = attn(24x24, padded 32x32) @ v̂  — one WMMA K-step (K=32).
// Output is written PIXEL-MAJOR: aot[b][pixel][c] (bf16) so each lane's 8
// accumulator rows are one contiguous 16-byte global_store_b128.
// ---------------------------------------------------------------------------
__global__ void k_attnv(const unsigned short* __restrict__ qkv0,
                        const float* __restrict__ wqd,
                        const unsigned short* __restrict__ attn,
                        unsigned short* __restrict__ aot) {
  constexpr int NC = 256, VSTR = 40;          // 80 B row stride (16B aligned)
  __shared__ __align__(16) unsigned short vs[NC][VSTR];   // [pixel][d]
  const int blk = blockIdx.x;                 // b*512 + h*64 + chunk
  const int chunk = blk & 63;
  const int h = (blk >> 6) & 7;
  const int b = blk >> 9;
  const int tid = threadIdx.x;
  const int wave = tid >> 5, lane = tid & 31;
  const int lhi = lane >> 4, llo = lane & 15;

  const int n = chunk * NC + tid;
  const int y = n >> 7, xx = n & 127;
  for (int c = 0; c < CHEAD; ++c) {
    const int vc = 2 * C_N + h * CHEAD + c;
    const unsigned short* vp = qkv0 + ((size_t)b * C3_N + vc) * NPIX;
    const float* wv = wqd + vc * 9;
    float av = 0.f;
#pragma unroll
    for (int dy = -1; dy <= 1; ++dy) {
      int yy = y + dy; if ((unsigned)yy >= (unsigned)HGT) continue;
#pragma unroll
      for (int dx = -1; dx <= 1; ++dx) {
        int xc = xx + dx; if ((unsigned)xc >= (unsigned)WID) continue;
        av += bf2f(vp[yy * WID + xc]) * wv[(dy + 1) * 3 + (dx + 1)];
      }
    }
    vs[tid][c] = f2bf(av);
  }
#pragma unroll
  for (int c = CHEAD; c < 32; ++c) vs[tid][c] = 0;   // zero-pad d=24..31
  __syncthreads();

  const unsigned short* ap = attn + (size_t)(b * NHEADS + h) * 1024;
  BF16Frag a[2];
#pragma unroll
  for (int t = 0; t < 2; ++t) {
    const int m = t * 16 + llo;
    a[t].q[0] = *reinterpret_cast<const v4u*>(&ap[m * 32 + lhi * 16]);
    a[t].q[1] = *reinterpret_cast<const v4u*>(&ap[m * 32 + lhi * 16 + 8]);
  }
  const v8f vz = {0.f,0.f,0.f,0.f,0.f,0.f,0.f,0.f};
#pragma unroll
  for (int t = 0; t < 2; ++t) {
    const int n_off = (wave * 2 + t) * 16;
    BF16Frag bb;
    const v4u* bp = reinterpret_cast<const v4u*>(&vs[n_off + llo][lhi * 16]);
    bb.q[0] = bp[0];
    bb.q[1] = bp[1];
    v8f acc0 = __builtin_amdgcn_wmma_f32_16x16x32_bf16(false, a[0].v, false, bb.v, (short)0, vz, false, false);
    v8f acc1 = __builtin_amdgcn_wmma_f32_16x16x32_bf16(false, a[1].v, false, bb.v, (short)0, vz, false, false);
    const size_t col = (size_t)chunk * NC + n_off + llo;
    unsigned short* rowp = aot + ((size_t)b * NPIX + col) * C_N + h * CHEAD;
    // acc0 rows: c = 8*lhi .. 8*lhi+7 ; acc1 rows: c = 16..23 (lhi==0 only)
    *reinterpret_cast<v4u*>(rowp + 8 * lhi) = pack8bf(acc0);
    if (lhi == 0)
      *reinterpret_cast<v4u*>(rowp + 16) = pack8bf(acc1);
  }
}

// ---------------------------------------------------------------------------
// Kernel 7: output 1x1 projection GEMM reading pixel-major bf16 aot.
// B tile (128 pixels x 32 ch = straight 2D copy) staged with
// GLOBAL_LOAD_ASYNC_TO_LDS_B128 (ASYNCcnt), overlapped with A staging.
// ---------------------------------------------------------------------------
__global__ void k_gemm_proj(const float* __restrict__ Wt,              // [192][192]
                            const unsigned short* __restrict__ aot,    // [B][NPIX][C_N]
                            float* __restrict__ Yout) {
  constexpr int BM = 64, BN = 128, BK = 32, K = C_N;
  __shared__ __align__(16) unsigned short As[BM][BK + 8];   // [m][k], 80B stride
  __shared__ __align__(16) unsigned short Bs[BN][BK + 8];   // [n][k], 80B stride
  const int b = blockIdx.z;
  const unsigned short* X = aot + (size_t)b * NPIX * C_N;
  float* Y = Yout + (size_t)b * (size_t)C_N * NPIX;
  const int m_base = blockIdx.y * BM;
  const int n_base = blockIdx.x * BN;
  const int tid = threadIdx.x;
  const int wave = tid >> 5, lane = tid & 31;
  const int wm = (wave >> 2) * 32;
  const int wn = (wave & 3) * 32;
  const int lhi = lane >> 4, llo = lane & 15;

  const unsigned long long gbase = (unsigned long long)(size_t)X;

  const v8f vz = {0.f,0.f,0.f,0.f,0.f,0.f,0.f,0.f};
  v8f acc[2][2];
  acc[0][0] = vz; acc[0][1] = vz; acc[1][0] = vz; acc[1][1] = vz;

  // per-thread async-copy assignments: 512 x 16B chunks, 2 per thread
  const int r0 = tid >> 2,           c0 = tid & 3;
  const int r1 = (tid + 256) >> 2,   c1 = (tid + 256) & 3;
  const unsigned lds0 = (unsigned)(size_t)&Bs[r0][c0 * 8];
  const unsigned lds1 = (unsigned)(size_t)&Bs[r1][c1 * 8];

  for (int kb = 0; kb < K; kb += BK) {
    // issue async global->LDS copies of the B tile (bf16, no conversion)
    {
      unsigned goff0 = (unsigned)(((unsigned)(n_base + r0) * C_N + kb + c0 * 8) * 2);
      unsigned goff1 = (unsigned)(((unsigned)(n_base + r1) * C_N + kb + c1 * 8) * 2);
      asm volatile(
          "global_load_async_to_lds_b128 %0, %1, %4 offset:0\n\t"
          "global_load_async_to_lds_b128 %2, %3, %4 offset:0"
          :: "v"(lds0), "v"(goff0), "v"(lds1), "v"(goff1), "s"(gbase)
          : "memory");
    }
    // overlap: stage A tile (f32 weights -> bf16)
    for (int j = tid; j < BM * BK; j += 256) {
      int r = j >> 5, cc = j & 31;
      As[r][cc] = f2bf(Wt[(size_t)(m_base + r) * K + kb + cc]);
    }
    asm volatile("s_wait_asynccnt 0x0" ::: "memory");
    __syncthreads();

    BF16Frag a[2], bb[2];
#pragma unroll
    for (int t = 0; t < 2; ++t) {
      const int mrow = wm + t * 16 + llo;
      a[t].q[0] = *reinterpret_cast<const v4u*>(&As[mrow][lhi * 16]);
      a[t].q[1] = *reinterpret_cast<const v4u*>(&As[mrow][lhi * 16 + 8]);
      const int ncol = wn + t * 16 + llo;
      const v4u* bp = reinterpret_cast<const v4u*>(&Bs[ncol][lhi * 16]);
      bb[t].q[0] = bp[0];
      bb[t].q[1] = bp[1];
    }
#pragma unroll
    for (int i = 0; i < 2; ++i)
#pragma unroll
      for (int j = 0; j < 2; ++j)
        acc[i][j] = __builtin_amdgcn_wmma_f32_16x16x32_bf16(
            false, a[i].v, false, bb[j].v, (short)0, acc[i][j], false, false);
    __syncthreads();
  }
#pragma unroll
  for (int i = 0; i < 2; ++i)
#pragma unroll
    for (int j = 0; j < 2; ++j)
#pragma unroll
      for (int r = 0; r < 8; ++r) {
        int m = m_base + wm + i * 16 + r + 8 * lhi;
        int n = n_base + wn + j * 16 + llo;
        Y[(size_t)m * NPIX + n] = acc[i][j][r];
      }
}

// ---------------------------------------------------------------------------
extern "C" void kernel_launch(void* const* d_in, const int* in_sizes, int n_in,
                              void* d_out, int out_size, void* d_ws, size_t ws_size,
                              hipStream_t stream) {
  (void)in_sizes; (void)n_in; (void)out_size; (void)ws_size;
  const float* x           = (const float*)d_in[0];
  const float* w_dw        = (const float*)d_in[1];
  const float* w_qkv       = (const float*)d_in[2];
  const float* w_qkvdw     = (const float*)d_in[3];
  const float* w_proj      = (const float*)d_in[4];
  const float* temperature = (const float*)d_in[5];
  const float* rescale     = (const float*)d_in[6];

  char* ws = (char*)d_ws;
  unsigned short* qkv0 = (unsigned short*)ws; ws += (size_t)B_N * C3_N * NPIX * 2;   // 151 MB
  unsigned short* aot  = (unsigned short*)ws; ws += (size_t)B_N * NPIX * C_N * 2;    //  50 MB
  float* var  = (float*)ws; ws += (size_t)B_N * C_N * 4;
  float* invq = (float*)ws; ws += (size_t)B_N * C_N * 4;
  float* invk = (float*)ws; ws += (size_t)B_N * C_N * 4;
  unsigned short* attn = (unsigned short*)ws; ws += (size_t)B_N * NHEADS * 1024 * 2;
  float* gram_part = (float*)ws; ws += (size_t)B_N * NHEADS * GSEG * 1024 * 4;       //   2 MB

  k_dwvar<<<B_N * C_N, 256, 0, stream>>>(x, w_dw, var);
  k_gemm_wmma<float, unsigned short>
      <<<dim3(NPIX / 128, C3_N / 64, B_N), 256, 0, stream>>>(w_qkv, x, qkv0, C3_N, C_N);
  k_norm<<<B_N * 2 * C_N, 256, 0, stream>>>(qkv0, w_qkvdw, invq, invk);
  k_gram_part<<<B_N * NHEADS * GSEG, 256, 0, stream>>>(qkv0, w_qkvdw, gram_part);
  k_softmax<<<B_N * NHEADS, 256, 0, stream>>>(gram_part, invq, invk,
                                              temperature, rescale, var, attn);
  k_attnv<<<B_N * NHEADS * 64, 256, 0, stream>>>(qkv0, w_qkvdw, attn, aot);
  k_gemm_proj<<<dim3(NPIX / 128, C_N / 64, B_N), 256, 0, stream>>>(w_proj, aot, (float*)d_out);
}
